// CodecTransformer_44753559224663
// MI455X (gfx1250) — matrix operand order repaired and checked by reading, not compiled, not run on hardware
//
#include <hip/hip_runtime.h>
#include <hip/hip_bf16.h>

// ---------------- constants ----------------
#define D512   512
#define HEADS  8
#define HD     64
#define LAY    12
#define DFF    2048
#define DSPK   192
#define MAXLEN 200
#define TT     512
#define TK     712            // MAXLEN + T
#define BBATCH 4              // 2 cond + 2 uncond rows stacked
#define MROWS  (BBATCH*TT)    // 2048
#define VA_    1024
#define NA_    8
#define VB_    64
#define NB_    4
#define CFGW   1.5f

// GEMM tiling
#define BM 128
#define BN 128
#define BK 32
#define LDA 40   // halves per A-tile row (80B rows -> every 16B fragment chunk aligned)
#define LDB 40
#define ABUFB (BM*LDA*2)   // bytes per A LDS buffer
#define WBUFB (BN*LDB*2)   // bytes per W LDS buffer

typedef _Float16 v16h __attribute__((ext_vector_type(16)));
typedef _Float16 v8h  __attribute__((ext_vector_type(8)));
typedef float    v8f  __attribute__((ext_vector_type(8)));

enum { MODE_PLAIN = 0, MODE_KV = 1, MODE_HEADS = 2 };
#define FLAG_BIAS  1
#define FLAG_RES   2
#define FLAG_GELU  4
#define FLAG_OUTH  8   // store result as f16

// low 32 bits of a generic pointer into LDS = LDS byte offset (ISA 10.2 aperture rule)
#define LDSOFF(p) ((unsigned)(unsigned long long)(p))

__device__ __forceinline__ void async_b128(unsigned lds, const void* g)
{
    asm volatile("global_load_async_to_lds_b128 %0, %1, off"
                 :: "v"(lds), "v"(g) : "memory");
}
__device__ __forceinline__ void wait_async4()
{
    asm volatile("s_wait_asynccnt 0x4" ::: "memory");
}
__device__ __forceinline__ void wait_async0()
{
    asm volatile("s_wait_asynccnt 0x0" ::: "memory");
}

// ---------------- WMMA GEMM: out = epilogue(A[f16 MxK] @ Wt[f16 NxK]^T) ----------------
// A row-major [M][K] f16, Wt row-major [N][K] f16 (pre-transposed weights).
// Double-buffered LDS staging via global_load_async_to_lds_b128 (ASYNCcnt).
// MODE/FLAGS/FULLN are compile-time: straight-line async issue, branch-free epilogue.
template <int MODE, int FLAGS, bool FULLN>
__global__ __launch_bounds__(256)
void wmma_gemm_kernel(const _Float16* __restrict__ A, const _Float16* __restrict__ Wt,
                      const float* __restrict__ bias, const float* __restrict__ Res,
                      float* __restrict__ outC, float* __restrict__ outU,
                      int M, int N, int K, long wZStride, int biasZStride)
{
    __shared__ __align__(16) _Float16 As[2][BM * LDA];
    __shared__ __align__(16) _Float16 Ws[2][BN * LDB];

    const int tid   = threadIdx.x;
    const int wave  = tid >> 5;
    const int lane  = tid & 31;
    const int waveM = wave >> 1;      // 0..3 -> 32 rows each
    const int waveN = wave & 1;       // 0..1 -> 64 cols each
    const int l16   = lane & 15;
    const int lhi   = lane >> 4;
    const int blockN0 = blockIdx.x * BN;
    const int blockM0 = blockIdx.y * BM;
    const int z       = blockIdx.z;
    const _Float16* Wz = Wt + (long)z * wZStride;

    // per-thread staging assignment: 256 threads cover 128 rows x 4 16B-chunks, twice
    const int cm = tid >> 2;              // row 0..63 (second row = cm+64)
    const int ck = (tid & 3) * 8;         // half-offset within row: 0,8,16,24
    const _Float16* aP0 = A  + (long)(blockM0 + cm) * K + ck;
    const _Float16* aP1 = aP0 + (long)64 * K;
    const _Float16* wP0 = Wz + (long)(blockN0 + cm) * K + ck;
    const _Float16* wP1 = wP0 + (long)64 * K;
    const unsigned aL0 = LDSOFF(&As[0][cm * LDA + ck]);
    const unsigned aL1 = LDSOFF(&As[0][(cm + 64) * LDA + ck]);
    const unsigned wL0 = LDSOFF(&Ws[0][cm * LDB + ck]);
    const unsigned wL1 = LDSOFF(&Ws[0][(cm + 64) * LDB + ck]);
    const bool ok0 = FULLN || ((blockN0 + cm) < N);
    const bool ok1 = FULLN || ((blockN0 + cm + 64) < N);
    if (!FULLN) {                 // zero LDS rows that will never be async-written
        v8h z8 = {};
        if (!ok0) { *(v8h*)&Ws[0][cm * LDB + ck] = z8; *(v8h*)&Ws[1][cm * LDB + ck] = z8; }
        if (!ok1) { *(v8h*)&Ws[0][(cm + 64) * LDB + ck] = z8; *(v8h*)&Ws[1][(cm + 64) * LDB + ck] = z8; }
    }

    v8f acc[2][4];
    v8f zero = {};
#pragma unroll
    for (int i = 0; i < 2; ++i)
#pragma unroll
        for (int j = 0; j < 4; ++j) acc[i][j] = zero;

    // staging pointers advance by += BK each issue (no per-iter address rebuild)
    const _Float16 *a0 = aP0, *a1 = aP1, *w0 = wP0, *w1 = wP1;

#define ISSUE_TILE(bufofsA, bufofsW)                          \
    do {                                                      \
        async_b128(aL0 + (bufofsA), a0);                      \
        async_b128(aL1 + (bufofsA), a1);                      \
        if (ok0) async_b128(wL0 + (bufofsW), w0);             \
        if (ok1) async_b128(wL1 + (bufofsW), w1);             \
        a0 += BK; a1 += BK; w0 += BK; w1 += BK;               \
    } while (0)

#define COMPUTE_TILE(buf)                                                         \
    do {                                                                          \
        const _Float16* asb = As[buf];                                            \
        const _Float16* wsb = Ws[buf];                                            \
        v16h afrag[2];                                                            \
        _Pragma("unroll")                                                         \
        for (int i = 0; i < 2; ++i) {                                             \
            int m  = waveM * 32 + i * 16 + l16;                                   \
            int b0 = lhi * 8;                                                     \
            v8h* dst = (v8h*)&afrag[i];                                           \
            dst[0] = *(const v8h*)&asb[m * LDA + b0];                             \
            dst[1] = *(const v8h*)&asb[m * LDA + 16 + b0];                        \
        }                                                                         \
        v16h bfrag[4];                                                            \
        _Pragma("unroll")                                                         \
        for (int j = 0; j < 4; ++j) {                                             \
            int n  = waveN * 64 + j * 16 + l16;                                   \
            int kb = lhi * 8;                                                     \
            v8h* dst = (v8h*)&bfrag[j];                                           \
            const _Float16* src = &wsb[n * LDB];                                  \
            dst[0] = *(const v8h*)&src[kb];                                       \
            dst[1] = *(const v8h*)&src[16 + kb];                                  \
        }                                                                         \
        _Pragma("unroll")                                                         \
        for (int i = 0; i < 2; ++i)                                               \
            _Pragma("unroll")                                                     \
            for (int j = 0; j < 4; ++j)                                           \
                acc[i][j] = __builtin_amdgcn_wmma_f32_16x16x32_f16(               \
                    false, afrag[i], false, bfrag[j], (short)0, acc[i][j],        \
                    false, false);                                                \
    } while (0)

    const int nk = K / BK;          // >= 2 for all our shapes
    ISSUE_TILE(0u, 0u);
    for (int ki = 0; ki < nk - 1; ++ki) {
        const int buf = ki & 1;
        if (buf == 0) ISSUE_TILE((unsigned)ABUFB, (unsigned)WBUFB);
        else          ISSUE_TILE(0u, 0u);
        if (FULLN) wait_async4();   // in-order completion: the older tile has landed
        else       wait_async0();
        __syncthreads();
        COMPUTE_TILE(buf);
        __syncthreads();
    }
    wait_async0();
    __syncthreads();
    COMPUTE_TILE((nk - 1) & 1);
#undef ISSUE_TILE
#undef COMPUTE_TILE

    // epilogue: C/D layout (VGPR r, lane l) -> M = r + 8*(l>>4), N = l&15
#pragma unroll
    for (int i = 0; i < 2; ++i) {
#pragma unroll
        for (int j = 0; j < 4; ++j) {
#pragma unroll
            for (int r = 0; r < 8; ++r) {
                int row = blockM0 + waveM * 32 + i * 16 + r + lhi * 8;
                int col = blockN0 + waveN * 64 + j * 16 + l16;
                if (!FULLN && col >= N) continue;
                float v = acc[i][j][r];
                if (FLAGS & FLAG_BIAS) v += bias[(long)biasZStride * z + col];
                if (FLAGS & FLAG_GELU) v = 0.5f * v * (1.0f + erff(v * 0.70710678118654752f));
                if (FLAGS & FLAG_RES)  v += Res[(long)row * N + col];
                if (MODE == MODE_PLAIN) {
                    if (FLAGS & FLAG_OUTH) ((_Float16*)outC)[(long)row * N + col] = (_Float16)v;
                    else                   outC[(long)row * N + col] = v;
                } else if (MODE == MODE_KV) {
                    int bb = row >> 9, t = row & 511;
                    int h = col >> 6, hd = col & 63;
                    outC[((((long)bb * HEADS + h) * TK) + MAXLEN + t) * HD + hd] = v;
                } else { // MODE_HEADS: cond rows -> outC (ws), uncond rows -> outU (d_out)
                    int bb = row >> 9, t = row & 511;
                    float* p = outC; int b = bb;
                    if (bb >= 2) { p = outU; b = bb - 2; }
                    p[(((long)(b * gridDim.z + z)) * TT + t) * (long)N + col] = v;
                }
            }
        }
    }
}

// ---------------- weight f32->f16 transpose: dst[n][k] = (h)src[k][n] ----------------
__global__ __launch_bounds__(256)
void wtrans_kernel(const float* __restrict__ src, _Float16* __restrict__ dst, int K, int N)
{
    __shared__ _Float16 tile[32][33];
    const long zo = (long)blockIdx.z * K * N;
    const int n0 = blockIdx.x * 32, k0 = blockIdx.y * 32;
    const int tx = threadIdx.x & 31, ty = threadIdx.x >> 5;   // 32 x 8
#pragma unroll
    for (int r = 0; r < 4; ++r) {
        int k = k0 + ty + r * 8;
        tile[ty + r * 8][tx] = (_Float16)src[zo + (long)k * N + n0 + tx];   // coalesced in n
    }
    __syncthreads();
#pragma unroll
    for (int r = 0; r < 4; ++r) {
        int n = n0 + ty + r * 8;
        dst[zo + (long)n * K + k0 + tx] = tile[tx][ty + r * 8];            // coalesced in k
    }
}

// ---------------- LayerNorm: one wave per 512-wide row, f16 output ----------------
__global__ __launch_bounds__(256)
void ln_kernel(const float* __restrict__ x, const float* __restrict__ w,
               const float* __restrict__ b, _Float16* __restrict__ out)
{
    int row  = blockIdx.x * 8 + (threadIdx.x >> 5);
    int lane = threadIdx.x & 31;
    const float* xr = x + (long)row * D512;
    float v[16], s = 0.f, s2 = 0.f;
#pragma unroll
    for (int i = 0; i < 16; ++i) { v[i] = xr[lane + i * 32]; s += v[i]; s2 += v[i] * v[i]; }
#pragma unroll
    for (int o = 16; o >= 1; o >>= 1) { s += __shfl_xor(s, o, 32); s2 += __shfl_xor(s2, o, 32); }
    float m   = s * (1.0f / D512);
    float var = s2 * (1.0f / D512) - m * m;
    float rin = rsqrtf(var + 1e-5f);
    _Float16* orow = out + (long)row * D512;
#pragma unroll
    for (int i = 0; i < 16; ++i) { int d = lane + i * 32; orow[d] = (_Float16)((v[i] - m) * rin * w[d] + b[d]); }
}

// ---------------- speaker conditioning ----------------
__global__ void spk_kernel(const float* __restrict__ se, const float* __restrict__ Wspk,
                           const float* __restrict__ bspk, float* __restrict__ spk)
{
    int i = blockIdx.x * blockDim.x + threadIdx.x;
    if (i >= 2 * D512) return;
    int b = i >> 9, d = i & 511;
    float s = bspk[d];
    for (int k = 0; k < DSPK; ++k) s += se[b * DSPK + k] * Wspk[k * D512 + d];
    spk[i] = s;
}

// ---------------- stacked input X[4*512, 512]: rows 0..1023 cond, 1024..2047 uncond ----
__global__ void prep_kernel(const float* __restrict__ content, const float* __restrict__ state,
                            const float* __restrict__ spk, const float* __restrict__ bspk,
                            float* __restrict__ X)
{
    long i = (long)blockIdx.x * blockDim.x + threadIdx.x;
    if (i >= (long)MROWS * D512) return;
    int d  = (int)(i & 511);
    long r = i >> 9;
    int bb = (int)(r >> 9), t = (int)(r & 511);
    int b  = bb & 1;
    float cv = content[((long)(b * TT + t)) * D512 + d];
    X[i] = (bb < 2) ? (cv + state[((long)(b * TT + t)) * D512 + d] + spk[b * D512 + d])
                    : (cv + bspk[d]);
}

__global__ void zero_kernel(float* __restrict__ p, long n)
{
    long i = (long)blockIdx.x * blockDim.x + threadIdx.x;
    if (i < n) p[i] = 0.0f;
}

// ---------------- attention: one 64-thread block per (bb,h,q) row, f32 math, f16 out ----
__global__ __launch_bounds__(64)
void attn_kernel(const float* __restrict__ Q, const float* __restrict__ Kp,
                 const float* __restrict__ Vp, _Float16* __restrict__ O)
{
    int t = blockIdx.x, h = blockIdx.y, bb = blockIdx.z;
    int tid = threadIdx.x;
    __shared__ float q[HD];
    __shared__ float sc[TK];
    __shared__ float red[64];
    q[tid] = Q[(((long)bb * TT + t) * D512) + h * HD + tid];
    __syncthreads();
    int tkc = t + MAXLEN + 1;
    const float* Kb = Kp + (((long)bb * HEADS + h) * TK) * HD;
    float lmax = -1e30f;
    for (int k = tid; k < tkc; k += 64) {
        const float* kr = Kb + (long)k * HD;
        float s = 0.f;
#pragma unroll
        for (int d = 0; d < HD; ++d) s += q[d] * kr[d];
        s *= 0.125f;
        sc[k] = s;
        lmax = fmaxf(lmax, s);
    }
    red[tid] = lmax; __syncthreads();
    for (int o = 32; o >= 1; o >>= 1) { if (tid < o) red[tid] = fmaxf(red[tid], red[tid + o]); __syncthreads(); }
    float mx = red[0]; __syncthreads();
    float lsum = 0.f;
    for (int k = tid; k < tkc; k += 64) { float e = __expf(sc[k] - mx); sc[k] = e; lsum += e; }
    red[tid] = lsum; __syncthreads();
    for (int o = 32; o >= 1; o >>= 1) { if (tid < o) red[tid] += red[tid + o]; __syncthreads(); }
    float inv = 1.0f / red[0];
    const float* Vb = Vp + (((long)bb * HEADS + h) * TK) * HD;
    float acc = 0.f;
    for (int k = 0; k < tkc; ++k) acc += sc[k] * Vb[(long)k * HD + tid];
    O[(((long)bb * TT + t) * D512) + h * HD + tid] = (_Float16)(acc * inv);
}

// ---------------- copy this layer's cond-pass K,V into d_out kv region ----------------
__global__ void copykv_kernel(const float* __restrict__ Kp, const float* __restrict__ Vp,
                              float* __restrict__ dst, int layer)
{
    const long PART = (long)HEADS * TK * HD;   // 364544
    long i = (long)blockIdx.x * blockDim.x + threadIdx.x;
    if (i >= 4 * PART) return;
    long e = i % PART;
    int  q = (int)(i / PART);
    int  b = q >> 1, part = q & 1;
    float v = part ? Vp[(long)b * PART + e] : Kp[(long)b * PART + e];
    dst[(long)b * (LAY * 2 * PART) + (long)layer * (2 * PART) + (long)part * PART + e] = v;
}

// ---------------- CFG combine: u = u + CFG*(c - u) ----------------
__global__ void cfg_kernel(float* __restrict__ u, const float* __restrict__ c, long n)
{
    long i = (long)blockIdx.x * blockDim.x + threadIdx.x;
    if (i < n) { float uu = u[i]; u[i] = uu + CFGW * (c[i] - uu); }
}

// ---------------- host orchestration ----------------
extern "C" void kernel_launch(void* const* d_in, const int* in_sizes, int n_in,
                              void* d_out, int out_size, void* d_ws, size_t ws_size,
                              hipStream_t stream)
{
    (void)in_sizes; (void)n_in; (void)out_size; (void)ws_size;
    const float* content = (const float*)d_in[0];
    const float* state   = (const float*)d_in[1];
    const float* se      = (const float*)d_in[2];
    const float* Wspk    = (const float*)d_in[3];
    const float* bspk    = (const float*)d_in[4];
    const float* ln1_w   = (const float*)d_in[5];
    const float* ln1_b   = (const float*)d_in[6];
    const float* Wq      = (const float*)d_in[7];
    const float* bq      = (const float*)d_in[8];
    const float* Wk      = (const float*)d_in[9];
    const float* bk      = (const float*)d_in[10];
    const float* Wv      = (const float*)d_in[11];
    const float* bv      = (const float*)d_in[12];
    const float* Wo      = (const float*)d_in[13];
    const float* bo      = (const float*)d_in[14];
    const float* ln2_w   = (const float*)d_in[15];
    const float* ln2_b   = (const float*)d_in[16];
    const float* W1      = (const float*)d_in[17];
    const float* b1      = (const float*)d_in[18];
    const float* W2      = (const float*)d_in[19];
    const float* b2      = (const float*)d_in[20];
    const float* lnf_w   = (const float*)d_in[21];
    const float* lnf_b   = (const float*)d_in[22];
    const float* Wa      = (const float*)d_in[23];
    const float* ba      = (const float*)d_in[24];
    const float* Wb      = (const float*)d_in[25];
    const float* bb_     = (const float*)d_in[26];

    // ---- workspace layout ----
    float* ws = (float*)d_ws;
    long o = 0;
    float* SPK = ws + o; o += 1024;
    float* X   = ws + o; o += (long)MROWS * D512;          // f32 residual stream
    float* Qb  = ws + o; o += (long)MROWS * D512;          // f32 Q
    float* Kp  = ws + o; o += (long)BBATCH * HEADS * TK * HD;
    float* Vp  = ws + o; o += (long)BBATCH * HEADS * TK * HD;
    float* LAc = ws + o; o += (long)2 * NA_ * TT * VA_;
    float* LBc = ws + o; o += (long)2 * NB_ * TT * VB_;
    _Float16* hb = (_Float16*)(ws + o);                     // f16 region
    long ho = 0;
    _Float16* Hb16 = hb + ho; ho += (long)MROWS * D512;     // LN output
    _Float16* Ob16 = hb + ho; ho += (long)MROWS * D512;     // attention output
    _Float16* FF16 = hb + ho; ho += (long)MROWS * DFF;      // FFN hidden
    _Float16* WTq = hb + ho; ho += (long)LAY * D512 * D512;
    _Float16* WTk = hb + ho; ho += (long)LAY * D512 * D512;
    _Float16* WTv = hb + ho; ho += (long)LAY * D512 * D512;
    _Float16* WTo = hb + ho; ho += (long)LAY * D512 * D512;
    _Float16* WT1 = hb + ho; ho += (long)LAY * D512 * DFF;
    _Float16* WT2 = hb + ho; ho += (long)LAY * DFF * D512;
    _Float16* WTa = hb + ho; ho += (long)NA_ * D512 * VA_;
    _Float16* WTb = hb + ho; ho += (long)NB_ * D512 * VB_;

    float* out    = (float*)d_out;
    const long N_LA = (long)2 * NA_ * TT * VA_;
    const long N_LB = (long)2 * NB_ * TT * VB_;
    float* outLA = out;
    float* outLB = out + N_LA;
    float* outKV = out + N_LA + N_LB;

    // ---- one-time (per launch) weight transpose+convert to f16 [N][K] ----
    wtrans_kernel<<<dim3(16, 16, LAY), 256, 0, stream>>>(Wq, WTq, D512, D512);
    wtrans_kernel<<<dim3(16, 16, LAY), 256, 0, stream>>>(Wk, WTk, D512, D512);
    wtrans_kernel<<<dim3(16, 16, LAY), 256, 0, stream>>>(Wv, WTv, D512, D512);
    wtrans_kernel<<<dim3(16, 16, LAY), 256, 0, stream>>>(Wo, WTo, D512, D512);
    wtrans_kernel<<<dim3(64, 16, LAY), 256, 0, stream>>>(W1, WT1, D512, DFF);
    wtrans_kernel<<<dim3(16, 64, LAY), 256, 0, stream>>>(W2, WT2, DFF, D512);
    wtrans_kernel<<<dim3(32, 16, NA_), 256, 0, stream>>>(Wa, WTa, D512, VA_);
    wtrans_kernel<<<dim3(2, 16, NB_),  256, 0, stream>>>(Wb, WTb, D512, VB_);

    // ---- setup ----
    const long SZ_X  = (long)MROWS * D512;
    const long SZ_KV = (long)BBATCH * HEADS * TK * HD;
    spk_kernel<<<4, 256, 0, stream>>>(se, Wspk, bspk, SPK);
    prep_kernel<<<(int)((SZ_X + 255) / 256), 256, 0, stream>>>(content, state, SPK, bspk, X);
    zero_kernel<<<(int)((2 * SZ_KV + 255) / 256), 256, 0, stream>>>(Kp, 2 * SZ_KV);

    const dim3 gN512(4, 16, 1);
    const dim3 gN2048(16, 16, 1);
    for (int i = 0; i < LAY; ++i) {
        const long DD = (long)D512 * D512;
        ln_kernel<<<256, 256, 0, stream>>>(X, ln1_w + i * D512, ln1_b + i * D512, Hb16);
        wmma_gemm_kernel<MODE_PLAIN, FLAG_BIAS, true><<<gN512, 256, 0, stream>>>(
            Hb16, WTq + i * DD, bq + i * D512, nullptr, Qb, nullptr, MROWS, D512, D512, 0, 0);
        wmma_gemm_kernel<MODE_KV, FLAG_BIAS, true><<<gN512, 256, 0, stream>>>(
            Hb16, WTk + i * DD, bk + i * D512, nullptr, Kp, nullptr, MROWS, D512, D512, 0, 0);
        wmma_gemm_kernel<MODE_KV, FLAG_BIAS, true><<<gN512, 256, 0, stream>>>(
            Hb16, WTv + i * DD, bv + i * D512, nullptr, Vp, nullptr, MROWS, D512, D512, 0, 0);
        attn_kernel<<<dim3(TT, HEADS, BBATCH), 64, 0, stream>>>(Qb, Kp, Vp, Ob16);
        wmma_gemm_kernel<MODE_PLAIN, FLAG_BIAS | FLAG_RES, true><<<gN512, 256, 0, stream>>>(
            Ob16, WTo + i * DD, bo + i * D512, X, X, nullptr, MROWS, D512, D512, 0, 0);
        ln_kernel<<<256, 256, 0, stream>>>(X, ln2_w + i * D512, ln2_b + i * D512, Hb16);
        wmma_gemm_kernel<MODE_PLAIN, FLAG_BIAS | FLAG_GELU | FLAG_OUTH, true><<<gN2048, 256, 0, stream>>>(
            Hb16, WT1 + i * (long)D512 * DFF, b1 + i * DFF, nullptr,
            (float*)FF16, nullptr, MROWS, DFF, D512, 0, 0);
        wmma_gemm_kernel<MODE_PLAIN, FLAG_BIAS | FLAG_RES, true><<<gN512, 256, 0, stream>>>(
            FF16, WT2 + i * (long)DFF * D512, b2 + i * D512, X, X, nullptr, MROWS, D512, DFF, 0, 0);
        copykv_kernel<<<(int)((4 * (long)HEADS * TK * HD + 255) / 256), 256, 0, stream>>>(Kp, Vp, outKV, i);
    }

    // ---- final LN + CFG heads ----
    ln_kernel<<<256, 256, 0, stream>>>(X, lnf_w, lnf_b, Hb16);
    wmma_gemm_kernel<MODE_HEADS, FLAG_BIAS, true><<<dim3(VA_ / BN, 16, NA_), 256, 0, stream>>>(
        Hb16, WTa, ba, nullptr, LAc, outLA, MROWS, VA_, D512, (long)D512 * VA_, VA_);
    wmma_gemm_kernel<MODE_HEADS, FLAG_BIAS, false><<<dim3(1, 16, NB_), 256, 0, stream>>>(
        Hb16, WTb, bb_, nullptr, LBc, outLB, MROWS, VB_, D512, (long)D512 * VB_, VB_);
    cfg_kernel<<<(int)((N_LA + 255) / 256), 256, 0, stream>>>(outLA, LAc, N_LA);
    cfg_kernel<<<(int)((N_LB + 255) / 256), 256, 0, stream>>>(outLB, LBc, N_LB);
}
